// QLoRABigNet_67869073211753
// MI455X (gfx1250) — matrix operand order repaired and checked by reading, not compiled, not run on hardware
//
#include <hip/hip_runtime.h>

// ---------------------------------------------------------------------------
// QLoRA BigNet forward on gfx1250 (CDNA5, wave32, WMMA f16 -> f32 accum)
// Software-pipelined: double-buffered LDS, next tile's global loads overlap
// current tile's v_wmma ops; one barrier per k-iteration.
// ---------------------------------------------------------------------------

#define DIMN 1024
#define ROWS 16384
#define NGROUPS 64          // DIMN / GROUP(16)
#define LORA_SCALE_F 0.8f

typedef __attribute__((ext_vector_type(16))) _Float16 v16h;
typedef __attribute__((ext_vector_type(8)))  float    v8f;
typedef __attribute__((ext_vector_type(4)))  unsigned int v4u;

union FragH  { v16h h; v4u u[2]; };
union Pack4h { _Float16 h[4]; unsigned long long ull; };

constexpr int BM = 128, BN = 128, BK = 32;
constexpr int KT = DIMN / BK;   // 32 k-tiles
constexpr int LDT = 40;         // padded LDS row stride in halfs (80B rows)

// --------------------------- fused dequant GEMM ----------------------------
// OUT[m,n] = relu?( X @ W^T + bias + (U @ LB^T) [+ RESID] )
// W[n,k] = (Q[n,k]*2/15 - 1) * NRM[n, k/16];  U already holds 0.8 * (X @ LA^T)
__global__ __launch_bounds__(256) void qlora_gemm_kernel(
    const float* __restrict__ X, const int* __restrict__ Q,
    const float* __restrict__ NRM, const float* __restrict__ BIAS,
    const float* __restrict__ U, const float* __restrict__ LB,
    const float* __restrict__ RESID, float* __restrict__ OUT, int relu)
{
  __shared__ _Float16 As[2][BM * LDT];   // 2 x 10 KB
  __shared__ _Float16 Bs[2][BN * LDT];   // 2 x 10 KB

  const int tid  = threadIdx.x;
  const int bm   = blockIdx.y * BM;
  const int bn   = blockIdx.x * BN;
  const int lane = tid & 31;
  const int wave = tid >> 5;
  const int wm   = (wave >> 2) * 64;  // 2 waves along M
  const int wn   = (wave & 3) * 32;   // 4 waves along N

  v8f acc[4][2] = {};                 // 4 M-subtiles x 2 N-subtiles of 16x16

  const int lrow = tid >> 3;          // 0..31
  const int lcol = (tid & 7) * 4;     // 0,4,...,28

  // in-flight global tile (registers)
  float4 ga[4];
  int4   gq[4];
  float  gn[4];

  auto loadGlobal = [&](int kt) {
#pragma unroll
    for (int p = 0; p < 4; ++p) {
      const int r = lrow + p * 32;
      ga[p] = *(const float4*)(X + (size_t)(bm + r) * DIMN + kt + lcol);
      gq[p] = *(const int4*)  (Q + (size_t)(bn + r) * DIMN + kt + lcol);
      gn[p] = NRM[(bn + r) * NGROUPS + ((kt + lcol) >> 4)];
    }
  };

  auto stageLDS = [&](int b) {
#pragma unroll
    for (int p = 0; p < 4; ++p) {
      const int r = lrow + p * 32;
      Pack4h pa;
      pa.h[0] = (_Float16)ga[p].x; pa.h[1] = (_Float16)ga[p].y;
      pa.h[2] = (_Float16)ga[p].z; pa.h[3] = (_Float16)ga[p].w;
      *(unsigned long long*)&As[b][r * LDT + lcol] = pa.ull;
      Pack4h pb;
      pb.h[0] = (_Float16)(((float)gq[p].x * (2.0f / 15.0f) - 1.0f) * gn[p]);
      pb.h[1] = (_Float16)(((float)gq[p].y * (2.0f / 15.0f) - 1.0f) * gn[p]);
      pb.h[2] = (_Float16)(((float)gq[p].z * (2.0f / 15.0f) - 1.0f) * gn[p]);
      pb.h[3] = (_Float16)(((float)gq[p].w * (2.0f / 15.0f) - 1.0f) * gn[p]);
      *(unsigned long long*)&Bs[b][r * LDT + lcol] = pb.ull;
    }
  };

  // prologue: tile 0 into LDS buffer 0
  loadGlobal(0);
  stageLDS(0);

  for (int t = 0; t < KT; ++t) {
    const int cur = t & 1;
    // issue next tile's global loads before the barrier so they are in
    // flight during this tile's WMMA work
    if (t + 1 < KT) loadGlobal((t + 1) * BK);

    __syncthreads();   // LDS[cur] stores (from previous iteration) visible

    // ---- B fragments: 32x16, lanes 0-15 K=0..15, lanes 16-31 K=16..31 ----
    FragH bf[2];
#pragma unroll
    for (int ni = 0; ni < 2; ++ni) {
      const int c  = wn + ni * 16 + (lane & 15);
      const int k0 = (lane >> 4) * 16;
      bf[ni].u[0] = *(const v4u*)&Bs[cur][c * LDT + k0];
      bf[ni].u[1] = *(const v4u*)&Bs[cur][c * LDT + k0 + 8];
    }
    // ---- A fragments: 16x32, lanes 0-15 K=0..15, lanes 16-31 K=16..31 ----
#pragma unroll
    for (int mi = 0; mi < 4; ++mi) {
      const int rr = wm + mi * 16 + (lane & 15);
      const int k0 = (lane >> 4) * 16;
      FragH af;
      af.u[0] = *(const v4u*)&As[cur][rr * LDT + k0];
      af.u[1] = *(const v4u*)&As[cur][rr * LDT + k0 + 8];
#pragma unroll
      for (int ni = 0; ni < 2; ++ni)
        acc[mi][ni] = __builtin_amdgcn_wmma_f32_16x16x32_f16(
            false, af.h, false, bf[ni].h, (short)0, acc[mi][ni], false, false);
    }

    // convert + stage next tile into the other LDS buffer (no barrier needed:
    // writes target LDS[1-cur]; next iteration's barrier orders them)
    if (t + 1 < KT) stageLDS(1 - cur);
  }

  // ---- epilogue: bias + LoRA + residual + ReLU ----
  const int cl = lane & 15;
  const int hm = (lane >> 4) * 8;
#pragma unroll
  for (int mi = 0; mi < 4; ++mi) {
#pragma unroll
    for (int vi = 0; vi < 8; ++vi) {
      const int row = bm + wm + mi * 16 + hm + vi;
      const float4 uv = *(const float4*)(U + (size_t)row * 4);
#pragma unroll
      for (int ni = 0; ni < 2; ++ni) {
        const int col = bn + wn + ni * 16 + cl;
        const float4 lbv = *(const float4*)(LB + (size_t)col * 4);
        float v = acc[mi][ni][vi] + BIAS[col]
                + uv.x * lbv.x + uv.y * lbv.y + uv.z * lbv.z + uv.w * lbv.w;
        if (RESID) v += RESID[(size_t)row * DIMN + col];
        if (relu)  v = fmaxf(v, 0.0f);
        OUT[(size_t)row * DIMN + col] = v;
      }
    }
  }
}

// --------------------- rank-4 LoRA projection u = 0.8 * X @ LA^T -----------
__global__ __launch_bounds__(256) void lora_u_kernel(
    const float* __restrict__ X, const float* __restrict__ LA,
    float* __restrict__ U)
{
  const int lane = threadIdx.x & 31;
  const int row  = blockIdx.x * 8 + (threadIdx.x >> 5);
  const float* xr = X + (size_t)row * DIMN;
  float a0 = 0.f, a1 = 0.f, a2 = 0.f, a3 = 0.f;
  for (int k = lane; k < DIMN; k += 32) {
    const float xv = xr[k];
    a0 += xv * LA[k];
    a1 += xv * LA[DIMN + k];
    a2 += xv * LA[2 * DIMN + k];
    a3 += xv * LA[3 * DIMN + k];
  }
#pragma unroll
  for (int off = 16; off > 0; off >>= 1) {
    a0 += __shfl_xor(a0, off, 32);
    a1 += __shfl_xor(a1, off, 32);
    a2 += __shfl_xor(a2, off, 32);
    a3 += __shfl_xor(a3, off, 32);
  }
  if (lane == 0) {
    float4 r;
    r.x = a0 * LORA_SCALE_F; r.y = a1 * LORA_SCALE_F;
    r.z = a2 * LORA_SCALE_F; r.w = a3 * LORA_SCALE_F;
    *(float4*)(U + (size_t)row * 4) = r;
  }
}

// ------------------------------- LayerNorm ---------------------------------
__global__ __launch_bounds__(256) void layernorm_kernel(
    const float* __restrict__ X, const float* __restrict__ G,
    const float* __restrict__ B, float* __restrict__ OUT)
{
  const int lane = threadIdx.x & 31;
  const int row  = blockIdx.x * 8 + (threadIdx.x >> 5);
  const float* xr = X + (size_t)row * DIMN;
  float v[32];
  float s = 0.f, s2 = 0.f;
#pragma unroll
  for (int i = 0; i < 32; ++i) {
    const float t = xr[lane + i * 32];
    v[i] = t; s += t; s2 += t * t;
  }
#pragma unroll
  for (int off = 16; off > 0; off >>= 1) {
    s  += __shfl_xor(s,  off, 32);
    s2 += __shfl_xor(s2, off, 32);
  }
  const float mu  = s * (1.0f / DIMN);
  const float var = s2 * (1.0f / DIMN) - mu * mu;
  const float inv = rsqrtf(var + 1e-5f);
#pragma unroll
  for (int i = 0; i < 32; ++i) {
    const int c = lane + i * 32;
    OUT[(size_t)row * DIMN + c] = (v[i] - mu) * inv * G[c] + B[c];
  }
}

// ------------------------------- launcher ----------------------------------
extern "C" void kernel_launch(void* const* d_in, const int* in_sizes, int n_in,
                              void* d_out, int out_size, void* d_ws, size_t ws_size,
                              hipStream_t stream) {
  (void)in_sizes; (void)n_in; (void)out_size; (void)ws_size;
  const float* x     = (const float*)d_in[0];
  const int*   qw    = (const int*)  d_in[1];
  const float* wnorm = (const float*)d_in[2];
  const float* bias  = (const float*)d_in[3];
  const float* la    = (const float*)d_in[4];
  const float* lb    = (const float*)d_in[5];
  const float* lng   = (const float*)d_in[6];
  const float* lnb   = (const float*)d_in[7];
  float* out = (float*)d_out;

  const size_t MAT = (size_t)ROWS * DIMN;
  float* buf[3];
  buf[0] = (float*)d_ws;
  buf[1] = buf[0] + MAT;
  buf[2] = buf[1] + MAT;
  float* u = buf[2] + MAT;                    // ROWS * 4 floats

  const dim3 gGemm(DIMN / BN, ROWS / BM);     // 8 x 128 workgroups
  const dim3 gRow(ROWS / 8);                  // 8 rows per 256-thread block

  auto layer = [&](const float* in, int i, const float* resid, float* o, int relu) {
    lora_u_kernel<<<gRow, 256, 0, stream>>>(in, la + (size_t)i * 4 * DIMN, u);
    qlora_gemm_kernel<<<gGemm, 256, 0, stream>>>(
        in, qw + (size_t)i * DIMN * DIMN, wnorm + (size_t)i * DIMN * NGROUPS,
        bias + (size_t)i * DIMN, u, lb + (size_t)i * DIMN * 4, resid, o, relu);
  };

  const float* h = x;
  for (int blk = 0; blk < 6; ++blk) {
    // pick two scratch buffers distinct from the live residual pointer h
    float* A = nullptr; float* B = nullptr;
    for (int j = 0; j < 3; ++j) {
      if (buf[j] == h) continue;
      if (!A) A = buf[j];
      else if (!B) B = buf[j];
    }
    const int i0 = blk * 3;
    layer(h, i0,     nullptr, A, 1);          // linear + ReLU
    layer(A, i0 + 1, nullptr, B, 1);          // linear + ReLU
    float* C = (blk == 5) ? out : A;          // A is dead after layer 2
    layer(B, i0 + 2, h, C, 0);                // linear + residual
    if (blk < 5) {
      layernorm_kernel<<<gRow, 256, 0, stream>>>(
          C, lng + (size_t)blk * DIMN, lnb + (size_t)blk * DIMN, B);
      h = B;
    }
  }
}
